// sts_attention_59691455479831
// MI455X (gfx1250) — matrix operand order repaired and checked by reading, not compiled, not run on hardware
//
#include <hip/hip_runtime.h>

#define DEV __device__ __forceinline__

typedef __attribute__((ext_vector_type(16))) __bf16 v16bf;
typedef __attribute__((ext_vector_type(8)))  float  v8f;

union FragBF { v16bf v; unsigned short u[16]; unsigned int d[8]; };

#define NV   25
#define TT   128
#define P32  3200      // T*V
#define NPT  102400    // N*T*V
#define NELEM 6553600  // N*64*T*V

DEV unsigned short f2bf(float f) {
  union { float f; unsigned int u; } c; c.f = f;
  unsigned int r = (c.u + 0x7fffu + ((c.u >> 16) & 1u)) >> 16;
  return (unsigned short)r;
}

// A-matrix (16x32 bf16) K index of the even element of dword i (pair k,k+1)
DEV int a_kpair(int lane, int i) {
  int khalf = (lane < 16) ? 0 : 16;
  return khalf + 2 * i;
}
// B-matrix (32x16 bf16) K index of the even element of dword i
DEV int b_kpair(int lane, int i) { return ((lane < 16) ? 0 : 16) + 2 * i; }

DEV v8f wmma_bf16(const FragBF& a, const FragBF& b, v8f c) {
  return __builtin_amdgcn_wmma_f32_16x16x32_bf16(false, a.v, false, b.v,
                                                 (short)0, c, false, false);
}

struct P4 { const float* w0; const float* w1; const float* w2; const float* w3;
            const float* b0; const float* b1; const float* b2; const float* b3; };

// ---------------------------------------------------------------- xbar = mean_t(x)
__global__ void k_xbar(const float* __restrict__ x, float* __restrict__ xbar) {
  int nc = blockIdx.x;            // n*64+c
  int v = threadIdx.x;
  if (v >= NV) return;
  float s = 0.f;
  for (int t = 0; t < TT; ++t) s += x[((size_t)nc * TT + t) * NV + v];
  xbar[(size_t)nc * NV + v] = s * (1.0f / TT);
}

// ---------------------------------------------------------------- xT[n][p][c] = bf16(x[n][c][p])
__global__ void k_xT(const float* __restrict__ x, unsigned short* __restrict__ xT) {
  int idx = blockIdx.x * 256 + threadIdx.x;
  if (idx >= NELEM) return;
  int n = idx / (P32 * 64);
  int rem = idx % (P32 * 64);
  int p = rem >> 6, c = rem & 63;
  xT[idx] = f2bf(x[((size_t)(n * 64 + c)) * P32 + p]);
}

// ---------------------------------------------------------------- attention tensor -> bf16 B-layout
// attb[n][c][u(32)][k=h*32+w (96)]
__global__ void k_attn(const float* __restrict__ xbar,
                       const float* __restrict__ WKw, const float* __restrict__ WKb,
                       const float* __restrict__ WQw, const float* __restrict__ WQb,
                       const float* __restrict__ ffnw, const float* __restrict__ ffnb,
                       const float* __restrict__ attA, const float* __restrict__ attB,
                       const float* __restrict__ alpha, const float* __restrict__ adjc,
                       unsigned short* __restrict__ attb) {
  __shared__ float xb[1600];
  __shared__ float kk[200], qq[200];
  __shared__ float ta[5000];
  const int h = blockIdx.x % 3, n = blockIdx.x / 3;
  const int tid = threadIdx.x;

  for (int i = tid; i < 1600; i += 256) {
    int c = i / 25, vv = i % 25;
    xb[i] = xbar[((size_t)(n * 64 + c)) * NV + vv];
  }
  __syncthreads();
  for (int i = tid; i < 200; i += 256) {
    int r = i / 25, vv = i % 25;
    float sk = WKb[h * 8 + r], sq = WQb[h * 8 + r];
    for (int c = 0; c < 64; ++c) {
      float xv = xb[c * 25 + vv];
      sk += WKw[(h * 8 + r) * 64 + c] * xv;
      sq += WQw[(h * 8 + r) * 64 + c] * xv;
    }
    kk[i] = sk; qq[i] = sq;
  }
  __syncthreads();
  float aal = attA[h], abe = attB[h], al = alpha[h];
  for (int i = tid; i < 5000; i += 256) {
    int r = i / 625, rem = i % 625, ii = rem / 25, jj = rem % 25;
    float kv = kk[r * 25 + ii], qv = qq[r * 25 + jj];
    ta[i] = tanhf(aal * kv * qv + abe * (kv - qv));
  }
  __syncthreads();
  for (int i = tid; i < 64 * 32 * 32; i += 256) {
    int o = i >> 10, u = (i >> 5) & 31, w = i & 31;
    unsigned short val = 0;
    if (u < NV && w < NV) {
      float s = ffnb[h * 64 + o];
#pragma unroll
      for (int r = 0; r < 8; ++r) s += ffnw[(h * 64 + o) * 8 + r] * ta[r * 625 + u * 25 + w];
      val = f2bf(al * s + adjc[h * 625 + u * 25 + w]);
    }
    attb[((size_t)(n * 64 + o) * 32 + u) * 96 + h * 32 + w] = val;
  }
}

// ---------------------------------------------------------------- v = Wv @ x (+b)  [WMMA]
// vws[n][o][h][t][32] bf16 (w padded to 32; padding pre-zeroed)
// Each wave: 2 M-tiles (A frags 32 VGPR), strided column stripe -> no spills.
__global__ void k_vproj(const unsigned short* __restrict__ xT,
                        const float* __restrict__ WVw, const float* __restrict__ WVb,
                        unsigned short* __restrict__ vws) {
  const int hn = blockIdx.x, h = hn % 3, n = hn / 3;
  const int q = blockIdx.y;                        // 0..1
  const int lane = threadIdx.x & 31, wv = threadIdx.x >> 5;
  const int mhalf = wv & 1;                        // M tiles {0,1} or {2,3}
  const int grp = q * 4 + (wv >> 1);               // 8 column stripes

  FragBF a[2][2];
  float biasv[2][8];
  int   obase[2][8];
#pragma unroll
  for (int mm = 0; mm < 2; ++mm) {
    int mt = mhalf * 2 + mm;
#pragma unroll
    for (int kt = 0; kt < 2; ++kt)
#pragma unroll
      for (int i = 0; i < 8; ++i) {
        int row = mt * 16 + (lane & 15);
        int k = kt * 32 + a_kpair(lane, i);
        const float* wp = WVw + ((size_t)(h * 64 + row)) * 64 + k;
        a[mm][kt].u[2 * i]     = f2bf(wp[0]);
        a[mm][kt].u[2 * i + 1] = f2bf(wp[1]);
      }
#pragma unroll
    for (int g = 0; g < 8; ++g) {
      int o = mt * 16 + g + ((lane < 16) ? 0 : 8);
      biasv[mm][g] = WVb[h * 64 + o];
      obase[mm][g] = ((n * 64 + o) * 3 + h) * (TT * 32);
    }
  }

  const unsigned short* xrow = xT + (size_t)n * P32 * 64;
  for (int ct = grp; ct < 200; ct += 8) {
    int col = ct * 16 + (lane & 15);
    FragBF b[2];
#pragma unroll
    for (int kt = 0; kt < 2; ++kt)
#pragma unroll
      for (int i = 0; i < 8; ++i) {
        int k = kt * 32 + b_kpair(lane, i);
        b[kt].d[i] = *(const unsigned int*)(xrow + (size_t)col * 64 + k);
      }
    v8f z = {};
    v8f acc0 = z, acc1 = z;
#pragma unroll
    for (int kt = 0; kt < 2; ++kt) {
      acc0 = wmma_bf16(a[0][kt], b[kt], acc0);
      acc1 = wmma_bf16(a[1][kt], b[kt], acc1);
    }
    int t = col / NV, w = col - t * NV;
    int pw = t * 32 + w;
#pragma unroll
    for (int g = 0; g < 8; ++g) {
      vws[(size_t)obase[0][g] + pw] = f2bf(acc0[g] + biasv[0][g]);
      vws[(size_t)obase[1][g] + pw] = f2bf(acc1[g] + biasv[1][g]);
    }
  }
}

// ---------------------------------------------------------------- out[t,u] = v[t,(h,w)] @ att[(h,w),u]  [WMMA]
// B resident (48 VGPR), A streamed one fragment at a time.
__global__ void k_gcnmm(const unsigned short* __restrict__ vws,
                        const unsigned short* __restrict__ attb,
                        float* __restrict__ gcn_pre) {
  const int nc = blockIdx.x;                       // n*64+c
  const int lane = threadIdx.x & 31, wv = threadIdx.x >> 5; // 4 waves
  FragBF b[3][2];
#pragma unroll
  for (int kt = 0; kt < 3; ++kt)
#pragma unroll
    for (int nt = 0; nt < 2; ++nt)
#pragma unroll
      for (int i = 0; i < 8; ++i) {
        int u = nt * 16 + (lane & 15);
        int k = kt * 32 + b_kpair(lane, i);
        b[kt][nt].d[i] = *(const unsigned int*)(attb + ((size_t)nc * 32 + u) * 96 + k);
      }
#pragma unroll
  for (int mm = 0; mm < 2; ++mm) {
    int mt = wv * 2 + mm;                           // t-tile 0..7
    v8f z = {};
    v8f acc0 = z, acc1 = z;
#pragma unroll
    for (int kt = 0; kt < 3; ++kt) {
      FragBF a;
#pragma unroll
      for (int i = 0; i < 8; ++i) {
        int trow = mt * 16 + (lane & 15);
        int k = kt * 32 + a_kpair(lane, i);
        a.d[i] = *(const unsigned int*)(
            vws + (((size_t)nc * 3 + (k >> 5)) * TT + trow) * 32 + (k & 31));
      }
      acc0 = wmma_bf16(a, b[kt][0], acc0);
      acc1 = wmma_bf16(a, b[kt][1], acc1);
    }
#pragma unroll
    for (int nt = 0; nt < 2; ++nt) {
      int u = nt * 16 + (lane & 15);
      if (u < NV) {
        size_t base = (size_t)nc * P32 + u;
#pragma unroll
        for (int g = 0; g < 8; ++g) {
          int t = mt * 16 + g + ((lane < 16) ? 0 : 8);
          float val = (nt == 0) ? acc0[g] : acc1[g];
          gcn_pre[base + (size_t)t * NV] = val;
        }
      }
    }
  }
}

// ---------------------------------------------------------------- per-channel sum / sumsq
__global__ void k_stats(const float* __restrict__ buf, float* __restrict__ out) {
  __shared__ float s1[256], s2[256];
  int c = blockIdx.x;
  float sum = 0.f, sq = 0.f;
  for (int idx = threadIdx.x; idx < NPT; idx += 256) {
    int n = idx / P32, p = idx % P32;
    float v = buf[((size_t)(n * 64 + c)) * P32 + p];
    sum += v; sq += v * v;
  }
  s1[threadIdx.x] = sum; s2[threadIdx.x] = sq;
  __syncthreads();
  for (int s = 128; s > 0; s >>= 1) {
    if ((int)threadIdx.x < s) { s1[threadIdx.x] += s1[threadIdx.x + s]; s2[threadIdx.x] += s2[threadIdx.x + s]; }
    __syncthreads();
  }
  if (threadIdx.x == 0) { out[c] = s1[0]; out[64 + c] = s2[0]; }
}

// ---------------------------------------------------------------- BN + residual + relu + PE -> bf16 posmajor
__global__ void k_bnA(const float* __restrict__ gcn_pre, const float* __restrict__ st,
                      const float* __restrict__ gw, const float* __restrict__ gb,
                      const float* __restrict__ x, const float* __restrict__ pe,
                      unsigned short* __restrict__ ypeT) {
  int idx = blockIdx.x * 256 + threadIdx.x;
  if (idx >= NELEM) return;
  int c = idx & 63;
  int p = (idx >> 6) % P32;
  int n = idx / (P32 * 64);
  size_t src = ((size_t)(n * 64 + c)) * P32 + p;
  float m = st[c] * (1.0f / NPT);
  float var = st[64 + c] * (1.0f / NPT) - m * m;
  float y = (gcn_pre[src] - m) * rsqrtf(var + 1e-5f) * gw[c] + gb[c];
  y = fmaxf(y + x[src], 0.0f);
  y += pe[c * NV + (p % NV)];
  ypeT[idx] = f2bf(y);
}

// ---------------------------------------------------------------- TCN conv1 (all 4 branches fused) [WMMA]
__global__ void k_tcn1(const unsigned short* __restrict__ ypeT, P4 cw,
                       float* __restrict__ z1) {
  const int n = blockIdx.x, q = blockIdx.y;           // q 0..1
  const int lane = threadIdx.x & 31, wv = threadIdx.x >> 5;
  const int mhalf = wv & 1;
  const int grp = q * 4 + (wv >> 1);                  // 8 column stripes
  const float* wptr[4] = {cw.w0, cw.w1, cw.w2, cw.w3};
  const float* bptr[4] = {cw.b0, cw.b1, cw.b2, cw.b3};

  FragBF a[2][2];
  float biasv[2][8];
  int   obase[2][8];
#pragma unroll
  for (int mm = 0; mm < 2; ++mm) {
    int mt = mhalf * 2 + mm;                          // branch index == mt
#pragma unroll
    for (int kt = 0; kt < 2; ++kt)
#pragma unroll
      for (int i = 0; i < 8; ++i) {
        int oc = lane & 15;
        int k = kt * 32 + a_kpair(lane, i);
        const float* wp = (mt == 0 ? cw.w0 : mt == 1 ? cw.w1 : mt == 2 ? cw.w2 : cw.w3)
                          + oc * 64 + k;
        a[mm][kt].u[2 * i]     = f2bf(wp[0]);
        a[mm][kt].u[2 * i + 1] = f2bf(wp[1]);
      }
#pragma unroll
    for (int g = 0; g < 8; ++g) {
      int oc = g + ((lane < 16) ? 0 : 8);
      biasv[mm][g] = (mt == 0 ? cw.b0 : mt == 1 ? cw.b1 : mt == 2 ? cw.b2 : cw.b3)[oc];
      obase[mm][g] = (n * 64 + mt * 16 + oc) * P32;
    }
  }
  (void)wptr; (void)bptr;

  const unsigned short* yrow = ypeT + (size_t)n * P32 * 64;
  for (int ct = grp; ct < 200; ct += 8) {
    int col = ct * 16 + (lane & 15);
    FragBF b[2];
#pragma unroll
    for (int kt = 0; kt < 2; ++kt)
#pragma unroll
      for (int i = 0; i < 8; ++i) {
        int k = kt * 32 + b_kpair(lane, i);
        b[kt].d[i] = *(const unsigned int*)(yrow + (size_t)col * 64 + k);
      }
    v8f z = {};
    v8f acc0 = z, acc1 = z;
#pragma unroll
    for (int kt = 0; kt < 2; ++kt) {
      acc0 = wmma_bf16(a[0][kt], b[kt], acc0);
      acc1 = wmma_bf16(a[1][kt], b[kt], acc1);
    }
#pragma unroll
    for (int g = 0; g < 8; ++g) {
      z1[(size_t)obase[0][g] + col] = acc0[g] + biasv[0][g];
      z1[(size_t)obase[1][g] + col] = acc1[g] + biasv[1][g];
    }
  }
}

// ---------------------------------------------------------------- BN1 + relu -> bf16 posmajor
__global__ void k_bn1(const float* __restrict__ z1, const float* __restrict__ st,
                      P4 bn, unsigned short* __restrict__ z1r) {
  int idx = blockIdx.x * 256 + threadIdx.x;
  if (idx >= NELEM) return;
  int c = idx & 63;
  int p = (idx >> 6) % P32;
  int n = idx / (P32 * 64);
  int jb = c >> 4, cc = c & 15;
  const float* wp = (jb == 0) ? bn.w0 : (jb == 1) ? bn.w1 : (jb == 2) ? bn.w2 : bn.w3;
  const float* bp = (jb == 0) ? bn.b0 : (jb == 1) ? bn.b1 : (jb == 2) ? bn.b2 : bn.b3;
  size_t src = ((size_t)(n * 64 + c)) * P32 + p;
  float m = st[c] * (1.0f / NPT);
  float var = st[64 + c] * (1.0f / NPT) - m * m;
  float y = (z1[src] - m) * rsqrtf(var + 1e-5f) * wp[cc] + bp[cc];
  z1r[idx] = f2bf(fmaxf(y, 0.0f));
}

// ---------------------------------------------------------------- conv2 (3x3 dilated or 1x1) [WMMA im2col]
template <int NTAPS>
__global__ void k_conv2(const unsigned short* __restrict__ z1r,
                        const float* __restrict__ W, const float* __restrict__ bias,
                        float* __restrict__ z2, int j, int dil) {
  constexpr int KTOT = NTAPS * 16;
  constexpr int KT = (KTOT + 31) / 32;
  const int n = blockIdx.x, q = blockIdx.y;           // q in 0..1 -> 100 col tiles
  const int lane = threadIdx.x & 31, wv = threadIdx.x >> 5;

  FragBF a[KT];
#pragma unroll
  for (int kt = 0; kt < KT; ++kt)
#pragma unroll
    for (int i = 0; i < 8; ++i) {
      int oc = lane & 15;
      int k = kt * 32 + a_kpair(lane, i);
#pragma unroll
      for (int e = 0; e < 2; ++e) {
        int ke = k + e;
        unsigned short v = 0;
        if (ke < KTOT) {
          int tap = ke >> 4, ic = ke & 15;
          int widx = (NTAPS == 9) ? (((oc * 16 + ic) * 3 + tap / 3) * 3 + tap % 3)
                                  : (oc * 16 + ic);
          v = f2bf(W[widx]);
        }
        a[kt].u[2 * i + e] = v;
      }
    }
  float biasv[8];
  int   obase[8];
#pragma unroll
  for (int g = 0; g < 8; ++g) {
    int oc = g + ((lane < 16) ? 0 : 8);
    biasv[g] = bias[oc];
    obase[g] = (n * 64 + j * 16 + oc) * P32;
  }

  for (int ct = q * 100 + wv; ct < q * 100 + 100; ct += 8) {
    int col = ct * 16 + (lane & 15);
    int t = col / NV, v = col - t * NV;
    v8f acc = {};
#pragma unroll
    for (int kt = 0; kt < KT; ++kt) {
      FragBF b;
#pragma unroll
      for (int i = 0; i < 8; ++i) {
        int k = kt * 32 + b_kpair(lane, i);
        unsigned int dv = 0;
        if (k < KTOT) {
          int tap = k >> 4, ic = k & 15;
          int dt = (NTAPS == 9) ? (tap / 3 - 1) : 0;
          int dvv = (NTAPS == 9) ? (tap % 3 - 1) : 0;
          int ts = t + dil * dt, vs = v + dil * dvv;
          if (ts >= 0 && ts < TT && vs >= 0 && vs < NV)
            dv = *(const unsigned int*)(z1r + ((size_t)n * P32 + ts * NV + vs) * 64 + j * 16 + ic);
        }
        b.d[i] = dv;
      }
      acc = wmma_bf16(a[kt], b, acc);
    }
#pragma unroll
    for (int g = 0; g < 8; ++g)
      z2[(size_t)obase[g] + col] = acc[g] + biasv[g];
  }
}

// ---------------------------------------------------------------- BN2 + residual + relu -> output
__global__ void k_final(const float* __restrict__ z2, const float* __restrict__ st,
                        P4 bn, const float* __restrict__ x, float* __restrict__ out) {
  int idx = blockIdx.x * 256 + threadIdx.x;
  if (idx >= NELEM) return;
  int nc = idx / P32;
  int c = nc & 63;
  int jb = c >> 4, cc = c & 15;
  const float* wp = (jb == 0) ? bn.w0 : (jb == 1) ? bn.w1 : (jb == 2) ? bn.w2 : bn.w3;
  const float* bp = (jb == 0) ? bn.b0 : (jb == 1) ? bn.b1 : (jb == 2) ? bn.b2 : bn.b3;
  float m = st[c] * (1.0f / NPT);
  float var = st[64 + c] * (1.0f / NPT) - m * m;
  float y = (z2[idx] - m) * rsqrtf(var + 1e-5f) * wp[cc] + bp[cc];
  out[idx] = fmaxf(y + x[idx], 0.0f);
}

// ================================================================ host
extern "C" void kernel_launch(void* const* d_in, const int* in_sizes, int n_in,
                              void* d_out, int out_size, void* d_ws, size_t ws_size,
                              hipStream_t stream) {
  const float* x    = (const float*)d_in[0];
  const float* WKw  = (const float*)d_in[1];
  const float* WKb  = (const float*)d_in[2];
  const float* WQw  = (const float*)d_in[3];
  const float* WQb  = (const float*)d_in[4];
  const float* WVw  = (const float*)d_in[5];
  const float* WVb  = (const float*)d_in[6];
  const float* ffnw = (const float*)d_in[7];
  const float* ffnb = (const float*)d_in[8];
  const float* attA = (const float*)d_in[9];
  const float* attB = (const float*)d_in[10];
  const float* alpha= (const float*)d_in[11];
  const float* adjc = (const float*)d_in[12];
  const float* gbw  = (const float*)d_in[13];
  const float* gbb  = (const float*)d_in[14];
  const float* br[4][8];
  for (int j = 0; j < 4; ++j)
    for (int t = 0; t < 8; ++t) br[j][t] = (const float*)d_in[15 + 8 * j + t];
  const float* pe = (const float*)d_in[47];

  char* w = (char*)d_ws;
  size_t off = 0;
  auto alloc = [&](size_t bytes) {
    void* p = w + off;
    off = (off + bytes + 255) & ~(size_t)255;
    return p;
  };
  unsigned short* xT   = (unsigned short*)alloc((size_t)NELEM * 2);
  unsigned short* ypeT = (unsigned short*)alloc((size_t)NELEM * 2);
  unsigned short* z1r  = (unsigned short*)alloc((size_t)NELEM * 2);
  unsigned short* vws  = (unsigned short*)alloc((size_t)2048 * 3 * TT * 32 * 2);   // 50.3 MB
  unsigned short* attb = (unsigned short*)alloc((size_t)2048 * 32 * 96 * 2);       // 12.6 MB
  float* gcn_pre = (float*)alloc((size_t)NELEM * 4);
  float* z1      = (float*)alloc((size_t)NELEM * 4);
  float* z2      = (float*)alloc((size_t)NELEM * 4);
  float* xbar    = (float*)alloc((size_t)2048 * NV * 4);
  float* stA     = (float*)alloc(512);
  float* st1     = (float*)alloc(512);
  float* st2     = (float*)alloc(512);

  // zero v-workspace so the w=25..31 K padding contributes 0 in WMMA
  hipMemsetAsync(vws, 0, (size_t)2048 * 3 * TT * 32 * 2, stream);

  k_xbar<<<2048, 32, 0, stream>>>(x, xbar);
  k_xT<<<(NELEM + 255) / 256, 256, 0, stream>>>(x, xT);
  k_attn<<<96, 256, 0, stream>>>(xbar, WKw, WKb, WQw, WQb, ffnw, ffnb,
                                 attA, attB, alpha, adjc, attb);
  k_vproj<<<dim3(96, 2), 256, 0, stream>>>(xT, WVw, WVb, vws);
  k_gcnmm<<<2048, 128, 0, stream>>>(vws, attb, gcn_pre);
  k_stats<<<64, 256, 0, stream>>>(gcn_pre, stA);
  k_bnA<<<(NELEM + 255) / 256, 256, 0, stream>>>(gcn_pre, stA, gbw, gbb, x, pe, ypeT);

  P4 c1 = {br[0][0], br[1][0], br[2][0], br[3][0], br[0][1], br[1][1], br[2][1], br[3][1]};
  k_tcn1<<<dim3(32, 2), 256, 0, stream>>>(ypeT, c1, z1);
  k_stats<<<64, 256, 0, stream>>>(z1, st1);
  P4 bn1 = {br[0][2], br[1][2], br[2][2], br[3][2], br[0][3], br[1][3], br[2][3], br[3][3]};
  k_bn1<<<(NELEM + 255) / 256, 256, 0, stream>>>(z1, st1, bn1, z1r);

  k_conv2<9><<<dim3(32, 2), 256, 0, stream>>>(z1r, br[0][4], br[0][5], z2, 0, 1);
  k_conv2<9><<<dim3(32, 2), 256, 0, stream>>>(z1r, br[1][4], br[1][5], z2, 1, 2);
  k_conv2<1><<<dim3(32, 2), 256, 0, stream>>>(z1r, br[2][4], br[2][5], z2, 2, 1);
  k_conv2<1><<<dim3(32, 2), 256, 0, stream>>>(z1r, br[3][4], br[3][5], z2, 3, 1);

  k_stats<<<64, 256, 0, stream>>>(z2, st2);
  P4 bn2 = {br[0][6], br[1][6], br[2][6], br[3][6], br[0][7], br[1][7], br[2][7], br[3][7]};
  k_final<<<(NELEM + 255) / 256, 256, 0, stream>>>(z2, st2, bn2, x, (float*)d_out);
}
